// W2Attention_19198503813765
// MI455X (gfx1250) — matrix operand order repaired and checked by reading, not compiled, not run on hardware
//
#include <hip/hip_runtime.h>
#include <hip/hip_bf16.h>

// ---------------- CDNA5 WMMA / TDM types ----------------
typedef __attribute__((ext_vector_type(16))) _Float16 v16h;
typedef __attribute__((ext_vector_type(8)))  _Float16 v8h;
typedef __attribute__((ext_vector_type(8)))  float    v8f;
typedef __attribute__((ext_vector_type(4)))  unsigned int u32x4;
typedef __attribute__((ext_vector_type(8)))  int      i32x8;
typedef __attribute__((ext_vector_type(4)))  int      i32x4;

static constexpr int kS  = 2048;   // sequence length
static constexpr int kH  = 2048;   // hidden
static constexpr int kNH = 16;     // heads
static constexpr int kD  = 128;    // head dim
static constexpr int kN2 = 4096;   // 2*H (q/k projection width)

// LDS A-tile geometry for the TDM-staged GEMM: 64 rows x 32 halfs, padded by the
// TDM to a 40-half (80 B) row stride -> conflict-free ds_load_b128 fragments.
static constexpr int kAPad   = 40;            // halfs per padded row
static constexpr int kATileH = 2 * 64 * kAPad;// halfs for double buffer
static constexpr unsigned kATileBytes = 64 * kAPad * 2; // one buffer, bytes

__device__ __forceinline__ v16h make_v16(v8h lo, v8h hi) {
    return __builtin_shufflevector(lo, hi, 0,1,2,3,4,5,6,7,8,9,10,11,12,13,14,15);
}

// A-matrix 16x32 f16 fragment (ISA 7.12.2): lane m=lane&15, half=lane>>4.
// elems 0..7  = A[m][k0 + half*8 + j] ; elems 8..15 = A[m][k0 + 16 + half*8 + j]
__device__ __forceinline__ v16h load_a_frag(const _Float16* rowptr, int k0, int half) {
    const v8h* p0 = (const v8h*)(rowptr + k0 + half * 8);
    const v8h* p1 = (const v8h*)(rowptr + k0 + 16 + half * 8);
    return make_v16(*p0, *p1);
}

// B-matrix 32x16 f16 fragment: lane n=lane&15, half=lane>>4.
// elems 0..7 = B[k0 + half*8 + j][n]; elems 8..15 = B[k0 + 16 + half*8 + j][n]; rowptr = column n's K-contiguous storage.
__device__ __forceinline__ v16h load_b_frag(const _Float16* rowptr, int k0, int half) {
    const v8h* p0 = (const v8h*)(rowptr + k0 + half * 8);
    const v8h* p1 = (const v8h*)(rowptr + k0 + 16 + half * 8);
    return make_v16(*p0, *p1);
}

__device__ __forceinline__ v8f wmma_f16(v16h a, v16h b, v8f c) {
    return __builtin_amdgcn_wmma_f32_16x16x32_f16(false, a, false, b, (short)0, c, false, false);
}

// ---------------- Tensor Data Mover: stage a 64x32 f16 tile into LDS ----------------
// D# per CDNA5 ISA: group0 = {count/lds_addr/global_addr/type},
// group1 = {data_size, pad cfg, tensor_dim0/1, tile_dim0/1, dim0 stride}. 2D tensor:
// remaining groups zero. Padding: 4 dwords after every 16 dwords (one 64 B row)
// -> 80 B LDS row stride. This toolchain exposes the 6-arg builtin:
// (u32x4 g0, i32x8 g1, i32x4, i32x4, i32x8, i32 cpol).
__device__ __forceinline__ void tdm_load_tile_a(const _Float16* gbase, unsigned lds_off,
                                                int row0, int k0, int K, int M) {
    unsigned long long ga =
        (unsigned long long)(size_t)(gbase + (size_t)row0 * K + k0);
    u32x4 g0 = { 1u,                                   // count=1, user mode
                 lds_off,                              // lds_addr (bytes)
                 (unsigned)ga,                         // global_addr[31:0]
                 ((unsigned)(ga >> 32) & 0x01ffffffu) | (2u << 30) }; // [56:32] | type=2
    i32x8 g1 = { (int)((1u << 16) | (1u << 20) | (3u << 22) | (3u << 25)),
                 //     data_size=2B | pad_enable | pad_interval=16dw | pad_amount=4dw
                 (int)(((unsigned)K & 0xffffu) << 16),                     // tensor_dim0 lo
                 (int)((((unsigned)K >> 16) & 0xffffu) |
                       (((unsigned)M & 0xffffu) << 16)),                   // dim0 hi | dim1 lo
                 (int)((((unsigned)M >> 16) & 0xffffu) | (32u << 16)),     // dim1 hi | tile_dim0=32
                 64,                                                       // tile_dim1 = 64 rows
                 K,                                                        // tensor_dim0_stride lo
                 0, 0 };
    i32x4 gz4 = { 0, 0, 0, 0 };
    i32x8 gz8 = { 0, 0, 0, 0, 0, 0, 0, 0 };
    __builtin_amdgcn_tensor_load_to_lds(g0, g1, gz4, gz4, gz8, 0);
}

// ---------------- fp32 -> fp16 convert ----------------
__global__ void f32_to_f16_kernel(const float* __restrict__ in, _Float16* __restrict__ out, size_t n) {
    size_t i = (size_t)blockIdx.x * blockDim.x + threadIdx.x;
    if (i < n) out[i] = (_Float16)in[i];
}

// ---------------- GEMM: C[M,N] = A[M,K] * W[N,K]^T  (f16 in, f32 out) ----------------
// 4-wave workgroup computes 64(M) x 128(N). A tile (64x32) is TDM-staged into LDS
// (double buffered); each wave computes a 32x64 strip (2 A-frags x 4 B-frags, 8 WMMAs).
__global__ void __launch_bounds__(128)
gemm_xwt_f16(const _Float16* __restrict__ A, const _Float16* __restrict__ W,
             float* __restrict__ C, int M, int N, int K) {
    __shared__ alignas(16) _Float16 atile[kATileH];   // first (only) LDS alloc -> offset 0
    const int tid  = threadIdx.x;
    const int wave = tid >> 5, lane = tid & 31;
    const int r = lane & 15, hf = lane >> 4;
    const int m0 = blockIdx.y * 64;
    const int n0 = blockIdx.x * 128;
    const int mloc = (wave >> 1) * 32;                // this wave's rows inside the LDS tile
    const int nw = n0 + (wave & 1) * 64;              // this wave's first output column

    // The TDM writes atile behind the compiler's back (no IR-visible stores).
    // Escape atile's address into inline asm (input-only) with a memory clobber:
    // the optimizer must now assume asm/TDM can store to it, so fragment loads
    // can't be folded to undef — while accesses still go through the
    // addrspace(3) array and lower to ds_load_b128.
    {
        const _Float16* esc = atile;
        __asm__ volatile("" :: "v"(esc) : "memory");
    }

    const _Float16* bp[4];
#pragma unroll
    for (int t = 0; t < 4; ++t) bp[t] = W + (size_t)(nw + t * 16 + r) * K;

    v8f acc[2][4];
#pragma unroll
    for (int j = 0; j < 2; ++j)
#pragma unroll
        for (int t = 0; t < 4; ++t) { v8f z = {}; acc[j][t] = z; }

    if (wave == 0) tdm_load_tile_a(A, 0u, m0, 0, K, M);

    const int iters = K >> 5;
    for (int it = 0; it < iters; ++it) {
        const int k0  = it << 5;
        const int cur = it & 1;
        if (wave == 0) __builtin_amdgcn_s_wait_tensorcnt(0);
        __syncthreads();                               // LDS buf `cur` visible to all waves
        __asm__ volatile("" ::: "memory");             // TDM rewrote LDS: force re-load
        if (wave == 0 && it + 1 < iters)               // prior reads of buf cur^1 are done
            tdm_load_tile_a(A, (unsigned)(cur ^ 1) * kATileBytes, m0, k0 + 32, K, M);

        const _Float16* arow0 = atile + (size_t)cur * (64 * kAPad) + (size_t)(mloc + r) * kAPad;
        v16h a0 = load_a_frag(arow0, 0, hf);
        v16h a1 = load_a_frag(arow0 + 16 * kAPad, 0, hf);
#pragma unroll
        for (int t = 0; t < 4; ++t) {
            __builtin_prefetch(bp[t] + k0 + 256, 0, 1);   // global_prefetch down the K stream
            v16h b = load_b_frag(bp[t], k0, hf);
            acc[0][t] = wmma_f16(a0, b, acc[0][t]);
            acc[1][t] = wmma_f16(a1, b, acc[1][t]);
        }
    }

    // C/D layout: VGPR i -> row i (lanes 0-15) / row i+8 (lanes 16-31); col = lane&15
#pragma unroll
    for (int j = 0; j < 2; ++j)
#pragma unroll
        for (int t = 0; t < 4; ++t)
#pragma unroll
            for (int i = 0; i < 8; ++i) {
                int row = m0 + mloc + j * 16 + i + 8 * hf;
                C[(size_t)row * N + nw + t * 16 + r] = acc[j][t][i];
            }
}

// ---------------- post-processing: softplus / variance weighting / RoPE ----------------
__device__ __forceinline__ float softplus_f(float x) {
    return (x > 20.f) ? x : log1pf(__expf(x));
}

__device__ __forceinline__ float qk_weighted(const float* base, const float* cs,
                                             const float* sn, int s, int d, float* sg_out) {
    float mu  = base[d];
    float sg  = softplus_f(base[kD + d]) + 1e-4f;
    float w   = mu * rsqrtf(sg);
    int   dp  = (d < 64) ? d + 64 : d - 64;
    float mup = base[dp];
    float sgp = softplus_f(base[kD + dp]) + 1e-4f;
    float wp  = mup * rsqrtf(sgp);
    float rot = (d < 64) ? -wp : wp;
    *sg_out = sg;
    return w * cs[s * kD + d] + rot * sn[s * kD + d];
}

// q_f32 is [S, 2H]; head h occupies columns [h*2D, h*2D+2D): first D = mu, next D = sigma_raw.
__global__ void __launch_bounds__(128)
postproc_q(const float* __restrict__ qf, const float* __restrict__ cs,
           const float* __restrict__ sn, _Float16* __restrict__ qw) {
    int row = blockIdx.x;           // h * S + s
    int h = row >> 11, s = row & (kS - 1);
    int d = threadIdx.x;
    const float* base = qf + (size_t)s * kN2 + h * (2 * kD);
    float sg;
    float val = qk_weighted(base, cs, sn, s, d, &sg);
    qw[((size_t)h * kS + s) * kD + d] = (_Float16)val;
}

// Same, plus per-key bias c_k = 0.5*(sum k_w^2 + sum log sigma).
__global__ void __launch_bounds__(128)
postproc_k(const float* __restrict__ kf, const float* __restrict__ cs,
           const float* __restrict__ sn, _Float16* __restrict__ kw,
           float* __restrict__ ck) {
    __shared__ float red[128];
    int row = blockIdx.x;
    int h = row >> 11, s = row & (kS - 1);
    int d = threadIdx.x;
    const float* base = kf + (size_t)s * kN2 + h * (2 * kD);
    float sg;
    float val = qk_weighted(base, cs, sn, s, d, &sg);
    kw[((size_t)h * kS + s) * kD + d] = (_Float16)val;
    red[d] = val * val + logf(sg);
    __syncthreads();
#pragma unroll
    for (int off = 64; off > 0; off >>= 1) {
        if (d < off) red[d] += red[d + off];
        __syncthreads();
    }
    if (d == 0) ck[(size_t)h * kS + s] = 0.5f * red[0];
}

// vt[h][d][s] = (f16) v_f32[s][h*D + d]   (so PV B-fragments are K-contiguous)
__global__ void transpose_v(const float* __restrict__ vf, _Float16* __restrict__ vt) {
    size_t i = (size_t)blockIdx.x * blockDim.x + threadIdx.x;
    if (i >= (size_t)kNH * kD * kS) return;
    int s = (int)(i % kS);
    size_t hd = i / kS;
    int d = (int)(hd % kD);
    int h = (int)(hd / kD);
    vt[i] = (_Float16)vf[(size_t)s * kH + h * kD + d];
}

// ---------------- flash attention with per-key bias ----------------
// One wave per 16-query tile per head. Streams keys in chunks of 32.
__global__ void __launch_bounds__(32)
attn_kernel(const _Float16* __restrict__ qw, const _Float16* __restrict__ kw,
            const _Float16* __restrict__ vt, const float* __restrict__ ck,
            _Float16* __restrict__ attn /* [S,H] f16 */) {
    __shared__ _Float16 pbuf[16 * 32];   // P tile staging (C-layout -> A-layout)
    const int lane = threadIdx.x;
    const int r = lane & 15, hf = lane >> 4;
    const int head = blockIdx.x >> 7;            // S/16 = 128 q-tiles per head
    const int q0 = (blockIdx.x & 127) * 16;

    const _Float16* qh = qw + (size_t)head * kS * kD;
    const _Float16* kh = kw + (size_t)head * kS * kD;
    const _Float16* vh = vt + (size_t)head * kD * kS;
    const float*    ch = ck + (size_t)head * kS;

    v16h qa[4];
    const _Float16* qrow = qh + (size_t)(q0 + r) * kD;
#pragma unroll
    for (int c = 0; c < 4; ++c) qa[c] = load_a_frag(qrow, c * 32, hf);

    v8f o[8];
#pragma unroll
    for (int t = 0; t < 8; ++t) { v8f z = {}; o[t] = z; }
    float m_i[8], l_i[8];
#pragma unroll
    for (int i = 0; i < 8; ++i) { m_i[i] = -3.0e38f; l_i[i] = 0.f; }

    for (int k0 = 0; k0 < kS; k0 += 32) {
        // ---- scores: two 16x16 tiles (keys k0..k0+15, k0+16..k0+31) ----
        v8f s0 = {}, s1 = {};
        const _Float16* krow0 = kh + (size_t)(k0 + r) * kD;
        const _Float16* krow1 = kh + (size_t)(k0 + 16 + r) * kD;
#pragma unroll
        for (int c = 0; c < 4; ++c) {
            v16h b0 = load_b_frag(krow0, c * 32, hf);
            v16h b1 = load_b_frag(krow1, c * 32, hf);
            s0 = wmma_f16(qa[c], b0, s0);
            s1 = wmma_f16(qa[c], b1, s1);
        }
        // per-key bias (q-side terms cancel in softmax)
        float c0 = ch[k0 + r];
        float c1 = ch[k0 + 16 + r];
        float rm[8];
#pragma unroll
        for (int i = 0; i < 8; ++i) {
            s0[i] -= c0;
            s1[i] -= c1;
            rm[i] = fmaxf(s0[i], s1[i]);
        }
        // row max across the 16-lane half (C layout: col = lane&15)
#pragma unroll
        for (int off = 8; off > 0; off >>= 1)
#pragma unroll
            for (int i = 0; i < 8; ++i)
                rm[i] = fmaxf(rm[i], __shfl_xor(rm[i], off, 16));

        float pe0[8], pe1[8], rs[8];
#pragma unroll
        for (int i = 0; i < 8; ++i) {
            float mn = fmaxf(m_i[i], rm[i]);
            float sc = __expf(m_i[i] - mn);
            m_i[i] = mn;
            l_i[i] *= sc;
#pragma unroll
            for (int t = 0; t < 8; ++t) o[t][i] *= sc;
            pe0[i] = __expf(s0[i] - mn);
            pe1[i] = __expf(s1[i] - mn);
            rs[i] = pe0[i] + pe1[i];
        }
#pragma unroll
        for (int off = 8; off > 0; off >>= 1)
#pragma unroll
            for (int i = 0; i < 8; ++i)
                rs[i] += __shfl_xor(rs[i], off, 16);
#pragma unroll
        for (int i = 0; i < 8; ++i) l_i[i] += rs[i];

        // ---- P (16x32) C-layout -> LDS row-major -> A-fragment ----
#pragma unroll
        for (int i = 0; i < 8; ++i) {
            int row = i + 8 * hf;
            pbuf[row * 32 + r]      = (_Float16)pe0[i];
            pbuf[row * 32 + 16 + r] = (_Float16)pe1[i];
        }
        __asm__ volatile("" ::: "memory");   // same-wave LDS is in-order (DScnt)
        v16h pa = load_a_frag(pbuf + r * 32, 0, hf);
        __asm__ volatile("" ::: "memory");

        // ---- O += P * V : 8 d-tiles of 16 ----
#pragma unroll
        for (int t = 0; t < 8; ++t) {
            const _Float16* vrow = vh + (size_t)(t * 16 + r) * kS;
            v16h vb = load_b_frag(vrow, k0, hf);
            o[t] = wmma_f16(pa, vb, o[t]);
        }
    }

    // normalize and emit f16 [S,H]
#pragma unroll
    for (int i = 0; i < 8; ++i) {
        float inv = 1.0f / l_i[i];
        int row = q0 + i + 8 * hf;
#pragma unroll
        for (int t = 0; t < 8; ++t)
            attn[(size_t)row * kH + head * kD + t * 16 + r] = (_Float16)(o[t][i] * inv);
    }
}

// ---------------- host-side orchestration ----------------
extern "C" void kernel_launch(void* const* d_in, const int* in_sizes, int n_in,
                              void* d_out, int out_size, void* d_ws, size_t ws_size,
                              hipStream_t stream) {
    (void)in_sizes; (void)n_in; (void)out_size; (void)ws_size;
    const float* x  = (const float*)d_in[0];
    const float* cs = (const float*)d_in[1];
    const float* sn = (const float*)d_in[2];
    const float* wq = (const float*)d_in[3];
    const float* wk = (const float*)d_in[4];
    const float* wv = (const float*)d_in[5];
    const float* wo = (const float*)d_in[6];
    float* out = (float*)d_out;

    char* ws = (char*)d_ws;
    size_t off = 0;
    auto carve = [&](size_t bytes) -> void* {
        void* p = ws + off;
        off += (bytes + 255) & ~(size_t)255;
        return p;
    };
    _Float16* xh  = (_Float16*)carve((size_t)kS * kH * 2);
    _Float16* wqh = (_Float16*)carve((size_t)kN2 * kH * 2);
    _Float16* wkh = (_Float16*)carve((size_t)kN2 * kH * 2);
    _Float16* wvh = (_Float16*)carve((size_t)kH * kH * 2);
    _Float16* woh = (_Float16*)carve((size_t)kH * kH * 2);
    float*    qf  = (float*)   carve((size_t)kS * kN2 * 4);
    float*    kf  = (float*)   carve((size_t)kS * kN2 * 4);
    float*    vf  = (float*)   carve((size_t)kS * kH * 4);
    _Float16* qwh = (_Float16*)carve((size_t)kNH * kS * kD * 2);
    _Float16* kwh = (_Float16*)carve((size_t)kNH * kS * kD * 2);
    _Float16* vth = (_Float16*)carve((size_t)kNH * kD * kS * 2);
    float*    ckb = (float*)   carve((size_t)kNH * kS * 4);
    _Float16* ath = (_Float16*)carve((size_t)kS * kH * 2);

    auto cvt = [&](const float* src, _Float16* dst, size_t n) {
        f32_to_f16_kernel<<<dim3((unsigned)((n + 255) / 256)), dim3(256), 0, stream>>>(src, dst, n);
    };
    cvt(x,  xh,  (size_t)kS * kH);
    cvt(wq, wqh, (size_t)kN2 * kH);
    cvt(wk, wkh, (size_t)kN2 * kH);
    cvt(wv, wvh, (size_t)kH * kH);
    cvt(wo, woh, (size_t)kH * kH);

    // projections: y = x @ W^T   (grid = N/128 x M/64, 4 waves per block)
    gemm_xwt_f16<<<dim3(kN2 / 128, kS / 64), dim3(128), 0, stream>>>(xh, wqh, qf, kS, kN2, kH);
    gemm_xwt_f16<<<dim3(kN2 / 128, kS / 64), dim3(128), 0, stream>>>(xh, wkh, kf, kS, kN2, kH);
    gemm_xwt_f16<<<dim3(kH  / 128, kS / 64), dim3(128), 0, stream>>>(xh, wvh, vf, kS, kH,  kH);

    postproc_q<<<dim3(kNH * kS), dim3(128), 0, stream>>>(qf, cs, sn, qwh);
    postproc_k<<<dim3(kNH * kS), dim3(128), 0, stream>>>(kf, cs, sn, kwh, ckb);
    transpose_v<<<dim3((kNH * kD * kS + 255) / 256), dim3(256), 0, stream>>>(vf, vth);

    attn_kernel<<<dim3(kNH * (kS / 16)), dim3(32), 0, stream>>>(qwh, kwh, vth, ckb, ath);

    // output projection -> d_out (f32)
    gemm_xwt_f16<<<dim3(kH / 128, kS / 64), dim3(128), 0, stream>>>(ath, woh, out, kS, kH, kH);
}
